// KernelizedHeadAttention_42700564856981
// MI455X (gfx1250) — compile-verified, matrix operand order checked by reading, not run on hardware
//
#include <hip/hip_runtime.h>
#include <math.h>

// Problem constants (match reference)
#define B_    2
#define S_    2048
#define D_    1024
#define H_    16
#define DH_   64
#define DHID_ 128
#define DKER_ 64

typedef __attribute__((ext_vector_type(16))) _Float16 v16h;
typedef __attribute__((ext_vector_type(8)))  _Float16 v8h;
typedef __attribute__((ext_vector_type(8)))  float    v8f;

#define SHUF16(a,b) __builtin_shufflevector((a),(b),0,1,2,3,4,5,6,7,8,9,10,11,12,13,14,15)

__device__ __forceinline__ float gelu_exact(float x) {
  // jax.nn.gelu(approximate=False) = 0.5*x*(1+erf(x/sqrt(2)))
  return 0.5f * x * (1.0f + erff(x * 0.70710678118654752f));
}

// ---------------- Phase 1a: hid = gelu(xh @ W1), per-head 64 -> 128 ----------------
__global__ __launch_bounds__(128) void kha_feat1(const float* __restrict__ x,
                                                 const float* __restrict__ W1,
                                                 float* __restrict__ hid) {
  const int row = blockIdx.x;             // bh*S + s
  const int bh  = row >> 11;
  const int s   = row & (S_ - 1);
  const int b   = bh >> 4, h = bh & (H_ - 1);
  __shared__ float xs[DH_];
  const int e = threadIdx.x;              // hidden col
  if (e < DH_)
    xs[e] = x[((size_t)b * S_ + s) * D_ + h * DH_ + e];
  __syncthreads();
  const float* w = W1 + h * DH_ * DHID_ + e;   // W1[h][d][e]
  float acc = 0.f;
#pragma unroll
  for (int d = 0; d < DH_; ++d) acc = fmaf(xs[d], w[d * DHID_], acc);
  hid[(size_t)row * DHID_ + e] = gelu_exact(acc);
}

// ---------------- Phase 1b (Q): absqf = |gelu(hid @ Wq2)| as f16 ----------------
__global__ __launch_bounds__(64) void kha_feat2q(const float* __restrict__ hid,
                                                 const float* __restrict__ W2,
                                                 _Float16* __restrict__ absqf) {
  const int row = blockIdx.x;
  const int bh  = row >> 11;
  const int h   = bh & (H_ - 1);
  __shared__ float hs[DHID_];
  const int e = threadIdx.x;
  hs[e]      = hid[(size_t)row * DHID_ + e];
  hs[e + 64] = hid[(size_t)row * DHID_ + e + 64];
  __syncthreads();
  const float* w = W2 + h * DHID_ * DKER_ + e;
  float acc = 0.f;
#pragma unroll
  for (int f = 0; f < DHID_; ++f) acc = fmaf(hs[f], w[f * DKER_], acc);
  absqf[(size_t)row * DKER_ + e] = (_Float16)fabsf(gelu_exact(acc));
}

// ---------------- Phase 1b (K): abskf with scaling + interaction ----------------
__global__ __launch_bounds__(64) void kha_feat2k(const float* __restrict__ hid,
                                                 const float* __restrict__ W2,
                                                 const float* __restrict__ Wint,
                                                 const float* __restrict__ sD,
                                                 const float* __restrict__ sD2,
                                                 _Float16* __restrict__ abskf) {
  const int row = blockIdx.x;
  const int bh  = row >> 11;
  const int h   = bh & (H_ - 1);
  __shared__ float hs[DHID_];
  __shared__ float tt[DKER_];
  const int e = threadIdx.x;
  hs[e]      = hid[(size_t)row * DHID_ + e];
  hs[e + 64] = hid[(size_t)row * DHID_ + e + 64];
  __syncthreads();
  const float* w = W2 + h * DHID_ * DKER_ + e;
  float acc = 0.f;
#pragma unroll
  for (int f = 0; f < DHID_; ++f) acc = fmaf(hs[f], w[f * DKER_], acc);
  const float t = fabsf(sD[h * DKER_ + e]) * gelu_exact(acc);
  tt[e] = t;
  __syncthreads();
  const float* wi = Wint + h * DKER_ * DKER_ + e;
  float inter = 0.f;
#pragma unroll
  for (int f = 0; f < DKER_; ++f) inter = fmaf(tt[f], wi[f * DKER_], inter);
  const float fin = t + inter * sD2[h * DKER_ + e];
  abskf[(size_t)row * DKER_ + e] = (_Float16)fabsf(fin);
}

// ---------------- Phase 1c: vT[bh][d][t] = v[b][t][h*64+d] as f16 ----------------
__global__ __launch_bounds__(256) void kha_vtrans(const float* __restrict__ v,
                                                  _Float16* __restrict__ vT) {
  const int dr = blockIdx.y;              // bh*64 + d
  const int bh = dr >> 6;
  const int d  = dr & (DH_ - 1);
  const int b  = bh >> 4, h = bh & (H_ - 1);
  const int t  = blockIdx.x * 256 + threadIdx.x;
  vT[(size_t)dr * S_ + t] = (_Float16)v[((size_t)b * S_ + t) * D_ + h * DH_ + d];
}

// ---------------- Main fused kernel: one wave per 32-query tile ----------------
// grid = (S/128, B*H), block = 128 (4 waves); each wave owns 2 query groups of 16.
__global__ __launch_bounds__(128) void kha_main(const _Float16* __restrict__ absqf,
                                                const _Float16* __restrict__ abskf,
                                                const _Float16* __restrict__ vT,
                                                const unsigned char* __restrict__ mask,
                                                const float* __restrict__ slse,
                                                const float* __restrict__ sw,
                                                float* __restrict__ out) {
  const int bh   = blockIdx.y;
  const int b    = bh >> 4, h = bh & (H_ - 1);
  const int wave = threadIdx.x >> 5;
  const int lane = threadIdx.x & 31;
  const int lhi  = lane >> 4;             // half-wave select
  const int l15  = lane & 15;
  const int qbase = (blockIdx.x * 4 + wave) * 32;   // 32 queries per wave

  // B-operands: |qf| rows for 2 groups of 16 queries, K=64 -> two 16x32 operands each.
  v16h bq[2][2];
#pragma unroll
  for (int g = 0; g < 2; ++g) {
    const _Float16* qrow = absqf + ((size_t)bh * S_ + qbase + g * 16 + l15) * DKER_;
    v8h q0 = *(const v8h*)(qrow + lhi * 8);
    v8h q1 = *(const v8h*)(qrow + 16 + lhi * 8);
    v8h q2 = *(const v8h*)(qrow + 32 + lhi * 8);
    v8h q3 = *(const v8h*)(qrow + 48 + lhi * 8);
    bq[g][0] = SHUF16(q0, q1);
    bq[g][1] = SHUF16(q2, q3);
  }

  v8f oacc[2][4] = {{v8f{}, v8f{}, v8f{}, v8f{}}, {v8f{}, v8f{}, v8f{}, v8f{}}};
  float rowsum[2] = {0.f, 0.f};

  const unsigned char* mrow[2];
  const float* srow[2];
#pragma unroll
  for (int g = 0; g < 2; ++g) {
    mrow[g] = mask + ((size_t)bh * S_ + qbase + g * 16 + l15) * S_;
    srow[g] = sw   + ((size_t)bh * S_ + qbase + g * 16 + l15) * S_;
  }

  for (int t = 0; t < S_; t += 32) {
    v16h ap[2];   // P tiles in A-operand layout (row = query, K = 32 keys)
#pragma unroll
    for (int sub = 0; sub < 2; ++sub) {
      const int kb = t + sub * 16;
      // A-operand: |kf| rows for 16 keys (M = key), K = 64 features
      const _Float16* krow = abskf + ((size_t)bh * S_ + kb + l15) * DKER_;
      v8h p0 = *(const v8h*)(krow + lhi * 8);
      v8h p1 = *(const v8h*)(krow + 16 + lhi * 8);
      v8h p2 = *(const v8h*)(krow + 32 + lhi * 8);
      v8h p3 = *(const v8h*)(krow + 48 + lhi * 8);
      v16h ak0 = SHUF16(p0, p1);
      v16h ak1 = SHUF16(p2, p3);
#pragma unroll
      for (int g = 0; g < 2; ++g) {
        // S^T tile: c[r] = score(key = kb + r + 8*lhi, query = qbase + g*16 + l15)
        v8f c = {};
        c = __builtin_amdgcn_wmma_f32_16x16x32_f16(false, ak0, false, bq[g][0], (short)0, c, false, false);
        c = __builtin_amdgcn_wmma_f32_16x16x32_f16(false, ak1, false, bq[g][1], (short)0, c, false, false);

        // Branchless merge: a = mask ? score+1e-6 : exp(sparse_w); rowsum += mask ? score : 0
        const unsigned long long mb = *(const unsigned long long*)(mrow[g] + kb + lhi * 8);
        const float4* sp = (const float4*)(srow[g] + kb + lhi * 8);
        const float4 s0 = sp[0], s1 = sp[1];
        const float sv[8] = {s0.x, s0.y, s0.z, s0.w, s1.x, s1.y, s1.z, s1.w};
        float ev[8];
#pragma unroll
        for (int r = 0; r < 8; ++r) ev[r] = __expf(sv[r]);   // unconditional: no divergence
#pragma unroll
        for (int r = 0; r < 8; ++r) {
          const bool m   = ((mb >> (8 * r)) & 0xffull) != 0;
          const float sc = c[r];
          rowsum[g] += m ? sc : 0.0f;
          ap[g][sub * 8 + r] = (_Float16)(m ? sc + 1e-6f : ev[r]);  // lane-aligned C->A repack
        }
      }
    }
    // out tiles += P(16q x 32keys) @ V(32keys x 64dh): shared V B-operands
    const _Float16* vbase = vT + (size_t)bh * DH_ * S_ + t + lhi * 8;
#pragma unroll
    for (int j = 0; j < 4; ++j) {
      const _Float16* vrow = vbase + (size_t)(j * 16 + l15) * S_;
      v8h va = *(const v8h*)(vrow);
      v8h vb = *(const v8h*)(vrow + 16);
      v16h bv = SHUF16(va, vb);
#pragma unroll
      for (int g = 0; g < 2; ++g)
        oacc[g][j] = __builtin_amdgcn_wmma_f32_16x16x32_f16(false, ap[g], false, bv, (short)0,
                                                            oacc[g][j], false, false);
    }
  }

  // Normalize + store. Lanes L and L^16 hold complementary key-halves of rowsum.
#pragma unroll
  for (int g = 0; g < 2; ++g) {
    float rs = rowsum[g];
    rs += __int_as_float(__builtin_amdgcn_ds_bpermute((lane ^ 16) << 2, __float_as_int(rs)));
    const int qg  = qbase + g * 16 + l15;
    const float z = rs + 1e-6f + __expf(slse[(size_t)bh * S_ + qg]);
    const int invb = __float_as_int(1.0f / z);
    // Output C layout: lane col = dh (j*16 + l15), VGPR r = query (r + 8*lhi).
#pragma unroll
    for (int r = 0; r < 8; ++r) {
      const float zr = __int_as_float(
          __builtin_amdgcn_ds_bpermute((r + lhi * 8) << 2, invb));
      const int qo = qbase + g * 16 + r + lhi * 8;
      float* orow = out + ((size_t)b * S_ + qo) * D_ + h * DH_;
#pragma unroll
      for (int j = 0; j < 4; ++j)
        orow[j * 16 + l15] = oacc[g][j][r] * zr;
    }
  }
}

// ---------------- Launch ----------------
// Workspace layout (bytes):
//   [0, 32MiB)        hid   : f32 [B*H*S, 128]   (reused for Q then K)
//   [32MiB, +8MiB)    absqf : f16 [B*H*S, 64]
//   [+8MiB, +8MiB)    abskf : f16 [B*H*S, 64]
//   [+8MiB, +8MiB)    vT    : f16 [B*H, 64, S]
// total = 58,720,256 bytes
extern "C" void kernel_launch(void* const* d_in, const int* in_sizes, int n_in,
                              void* d_out, int out_size, void* d_ws, size_t ws_size,
                              hipStream_t stream) {
  const float* q    = (const float*)d_in[1];
  const float* k    = (const float*)d_in[2];
  const float* v    = (const float*)d_in[3];
  const unsigned char* mask = (const unsigned char*)d_in[4];   // bool mask
  const float* slse = (const float*)d_in[5];
  const float* sw   = (const float*)d_in[6];
  const float* Wq1  = (const float*)d_in[8];
  const float* Wk1  = (const float*)d_in[9];
  const float* Wq2  = (const float*)d_in[10];
  const float* Wk2  = (const float*)d_in[11];
  const float* Wint = (const float*)d_in[12];
  const float* sD   = (const float*)d_in[13];
  const float* sD2  = (const float*)d_in[14];

  char* ws = (char*)d_ws;
  float*    hid   = (float*)ws;
  _Float16* absqf = (_Float16*)(ws + (size_t)33554432);
  _Float16* abskf = (_Float16*)(ws + (size_t)33554432 + 8388608);
  _Float16* vT    = (_Float16*)(ws + (size_t)33554432 + 16777216);

  const int rows = B_ * H_ * S_;   // 65536

  // Q feature chain
  kha_feat1 <<<rows, 128, 0, stream>>>(q, Wq1, hid);
  kha_feat2q<<<rows,  64, 0, stream>>>(hid, Wq2, absqf);
  // K feature chain (reuses hid)
  kha_feat1 <<<rows, 128, 0, stream>>>(k, Wk1, hid);
  kha_feat2k<<<rows,  64, 0, stream>>>(hid, Wk2, Wint, sD, sD2, abskf);
  // V transpose to f16
  kha_vtrans<<<dim3(S_ / 256, B_ * H_ * DH_), 256, 0, stream>>>(v, vT);
  // Fused masked kernel-attention: 32 queries per wave
  kha_main<<<dim3(S_ / 128, B_ * H_), 128, 0, stream>>>(absqf, abskf, vT, mask, slse, sw,
                                                        (float*)d_out);
}
